// SNNReadout_63333587747464
// MI455X (gfx1250) — matrix operand order repaired and checked by reading, not compiled, not run on hardware
//
#include <hip/hip_runtime.h>

typedef __attribute__((ext_vector_type(16))) _Float16 v16h;
typedef __attribute__((ext_vector_type(8)))  _Float16 v8h;
typedef __attribute__((ext_vector_type(4)))  _Float16 v4h;
typedef __attribute__((ext_vector_type(8)))  float    v8f;
typedef __attribute__((ext_vector_type(4)))  float    v4f;

#define TSTEPS 100
#define BATCH  128
#define FDIM   4096
#define CDIM   10
#define MROWS  (TSTEPS * BATCH)   // 12800 rows of the flattened [T*B, F] GEMM
#define BETA   0.9f               // 1 - DT/TAU_M
#define VTH    1.0f
#define XSCALE 2.0f               // 1/(1-DROP_P), folded into W at staging

// ---------------------------------------------------------------------------
// GEMM partials: P[s, row, c] = sum over K-slice s of (spike*mask) @ (2W)^T
//   grid = (MROWS/64, S); block = 128 threads (4 waves, 16-row M-tile each)
//   KS = FDIM/S columns per split. W K-slice staged as f16 (pre-scaled by 2)
//   in LDS with:
//     - +8-half row pad -> B-fragment ds_load_b128 hits all 64 banks
//     - one extra all-zero row (index CDIM) that lanes 10..15 read via
//       same-address broadcast, so no per-iteration column-pad select.
// ---------------------------------------------------------------------------
template <int S>
__global__ __launch_bounds__(128) void snn_gemm_wmma(
    const float* __restrict__ spk,
    const float* __restrict__ msk,
    const float* __restrict__ W,
    float* __restrict__ P)
{
    constexpr int KS      = FDIM / S;   // K columns handled by this split
    constexpr int WSTRIDE = KS + 8;     // padded LDS row stride (halfs)
    constexpr int WROWS   = CDIM + 1;   // 10 real rows + 1 zero row
    __shared__ _Float16 Wl[WROWS * WSTRIDE];

    const int tid  = threadIdx.x;
    const int ks   = blockIdx.y;        // K-split index
    const int kOrg = ks * KS;           // global K origin of this split

    // Stage this split's W slice (f32 -> f16, pre-scaled by 1/keep) into LDS;
    // row CDIM is zero-filled padding for the N=16 WMMA tile.
    for (int i = tid * 4; i < WROWS * KS; i += 128 * 4) {
        const int c  = i / KS;          // KS is a power of two -> shift
        const int kl = i & (KS - 1);
        v4h h;
        if (c < CDIM) {
            v4f w = *(const v4f*)(W + c * FDIM + kOrg + kl);
            #pragma unroll
            for (int j = 0; j < 4; ++j) h[j] = (_Float16)(w[j] * XSCALE);
        } else {
            #pragma unroll
            for (int j = 0; j < 4; ++j) h[j] = (_Float16)0.f;
        }
        *(v4h*)(&Wl[c * WSTRIDE + kl]) = h;
    }
    __syncthreads();

    const int wave = tid >> 5;
    const int lane = tid & 31;
    const int m16  = lane & 15;         // row within tile (A) / column (B,C)
    const int hi   = lane >> 4;         // lane half-group

    const int rowBase   = blockIdx.x * 64 + wave * 16;
    const size_t rowOff = (size_t)(rowBase + m16) * FDIM + kOrg;
    const float* sRow = spk + rowOff;
    const float* mRow = msk + rowOff;

    // A fragment: lanes 0-15 -> K 0..7 & 16..23 ; lanes 16-31 -> +8
    const int kA0 = hi * 8;
    // B fragment: lanes 0-15 -> K 0..7 & 16..23 ; lanes 16-31 -> +8
    const int kB0 = hi * 8;

    const bool valid = (m16 < CDIM);
    const int  wrow  = valid ? m16 : CDIM;   // invalid lanes -> shared zero row
    const _Float16* wRow = &Wl[wrow * WSTRIDE];

    v8f acc = {};

    #pragma unroll 2
    for (int k = 0; k < KS; k += 32) {
        const v4f* sp0 = (const v4f*)(sRow + k + kA0);
        const v4f* mp0 = (const v4f*)(mRow + k + kA0);
        const v4f* sp1 = (const v4f*)(sRow + k + kA0 + 16);
        const v4f* mp1 = (const v4f*)(mRow + k + kA0 + 16);

        // streaming data: non-temporal b128 loads (read-once, keep L2 clean)
        v4f s0 = __builtin_nontemporal_load(sp0);
        v4f s1 = __builtin_nontemporal_load(sp0 + 1);
        v4f s2 = __builtin_nontemporal_load(sp1);
        v4f s3 = __builtin_nontemporal_load(sp1 + 1);
        v4f m0 = __builtin_nontemporal_load(mp0);
        v4f m1 = __builtin_nontemporal_load(mp0 + 1);
        v4f m2 = __builtin_nontemporal_load(mp1);
        v4f m3 = __builtin_nontemporal_load(mp1 + 1);

        // a = spike * mask (exact in f16: {0,1}); dropout scale lives in W
        v16h a;
        #pragma unroll
        for (int j = 0; j < 4; ++j) {
            a[j]      = (_Float16)(s0[j] * m0[j]);
            a[4 + j]  = (_Float16)(s1[j] * m1[j]);
            a[8 + j]  = (_Float16)(s2[j] * m2[j]);
            a[12 + j] = (_Float16)(s3[j] * m3[j]);
        }

        // B fragment: two conflict-free ds_load_b128 (zero row for pad lanes)
        v8h w0 = *(const v8h*)(wRow + k + kB0);
        v8h w1 = *(const v8h*)(wRow + k + kB0 + 16);
        v16h bfr;
        #pragma unroll
        for (int j = 0; j < 8; ++j) {
            bfr[j]     = w0[j];
            bfr[8 + j] = w1[j];
        }

        // D = A x B + C   (16x16x32, f32 accumulate)
        acc = __builtin_amdgcn_wmma_f32_16x16x32_f16(
            /*neg_a=*/false, a, /*neg_b=*/false, bfr,
            /*c_mod=*/(short)0, acc, /*reuse_a=*/false, /*reuse_b=*/false);
    }

    // C/D layout: VGPR i -> row = hi*8 + i, col = m16. Store partial (no bias).
    if (valid) {
        #pragma unroll
        for (int i = 0; i < 8; ++i) {
            const int row = rowBase + hi * 8 + i;
            P[(ks * MROWS + row) * CDIM + m16] = acc[i];
        }
    }
}

// ---------------------------------------------------------------------------
// LIF scan: reduce S partials + bias, then sequential leaky integrate-and-fire
// over T for each of the B*C = 1280 channels. P may alias out when S == 1
// (read-before-write per element), so no __restrict__ here.
// ---------------------------------------------------------------------------
__global__ __launch_bounds__(256) void snn_lif_scan(
    const float* P, int S, const float* bias, float* out)
{
    const int j = blockIdx.x * 256 + threadIdx.x;
    if (j >= BATCH * CDIM) return;
    const int b = j / CDIM;
    const int c = j - b * CDIM;
    const float bv = bias[c];

    float v = 0.f;
    for (int t = 0; t < TSTEPS; ++t) {
        const int row = t * BATCH + b;
        float cur = bv;
        for (int s = 0; s < S; ++s)
            cur += P[(s * MROWS + row) * CDIM + c];
        v = v * BETA + cur;
        const float sp = (v - VTH) > 0.f ? 1.f : 0.f;
        v -= sp * VTH;
        out[row * CDIM + c] = sp;
    }
}

// ---------------------------------------------------------------------------
extern "C" void kernel_launch(void* const* d_in, const int* in_sizes, int n_in,
                              void* d_out, int out_size, void* d_ws, size_t ws_size,
                              hipStream_t stream)
{
    const float* spk  = (const float*)d_in[0];  // [T,B,F]
    const float* msk  = (const float*)d_in[1];  // [T,B,F]
    const float* W    = (const float*)d_in[2];  // [C,F]
    const float* bias = (const float*)d_in[3];  // [C]
    float* out = (float*)d_out;                 // [T,B,C]

    const size_t splitBytes = (size_t)4 * MROWS * CDIM * sizeof(float); // 2 MB

    if (ws_size >= splitBytes) {
        // split-K x4: 800 workgroups / 3200 waves -> saturate HBM
        float* P = (float*)d_ws;
        snn_gemm_wmma<4><<<dim3(MROWS / 64, 4), 128, 0, stream>>>(spk, msk, W, P);
        snn_lif_scan<<<(BATCH * CDIM + 255) / 256, 256, 0, stream>>>(P, 4, bias, out);
    } else {
        // fallback: single K pass, partials live in d_out (aliased, S=1)
        snn_gemm_wmma<1><<<dim3(MROWS / 64, 1), 128, 0, stream>>>(spk, msk, W, out);
        snn_lif_scan<<<(BATCH * CDIM + 255) / 256, 256, 0, stream>>>(out, 1, bias, out);
    }
}
